// HierarchicalTemporalCompressor_78211354460519
// MI455X (gfx1250) — compile-verified, matrix operand order checked by reading, not run on hardware
//
#include <hip/hip_runtime.h>

// ---------------------------------------------------------------------------
// CDNA5 (gfx1250) HierarchicalTemporalCompressor: bf16 WMMA GEMMs with
// double-buffered async global->LDS staging (ASYNCcnt path) and ds_load_b128
// fragment reads. Every level: C[m,n] = sum_k A[m,k]*W[n,k] (+bias,+resid,tanh).
// ---------------------------------------------------------------------------

typedef __bf16        v16bf __attribute__((ext_vector_type(16)));
typedef float         v8f   __attribute__((ext_vector_type(8)));
typedef unsigned int  v4u   __attribute__((ext_vector_type(4)));

union BfPack {
  v4u   u[2];   // u[0] -> fragment VGPRs 0..3, u[1] -> VGPRs 4..7
  v16bf v;
};

__device__ __forceinline__ __bf16 f2bf(float f) {
  unsigned int u = __builtin_bit_cast(unsigned int, f);
  u += 0x7FFFu + ((u >> 16) & 1u);            // round-to-nearest-even
  unsigned short h = (unsigned short)(u >> 16);
  return __builtin_bit_cast(__bf16, h);
}

__device__ __forceinline__ float bf2f(__bf16 b) {
  unsigned short h = __builtin_bit_cast(unsigned short, b);
  unsigned int u = ((unsigned int)h) << 16;
  return __builtin_bit_cast(float, u);
}

__global__ void f32_to_bf16_kernel(const float* __restrict__ in,
                                   __bf16* __restrict__ out, long n) {
  long i = (long)blockIdx.x * blockDim.x + threadIdx.x;
  long stride = (long)gridDim.x * blockDim.x;
  for (; i < n; i += stride) out[i] = f2bf(in[i]);
}

// 16B async copy global -> LDS, tracked by ASYNCcnt (ISA 15.18.3 op 98).
__device__ __forceinline__ void asyncCopy16(unsigned int ldsAddr,
                                            const __bf16* gaddr) {
  asm volatile("global_load_async_to_lds_b128 %0, %1, off"
               :: "v"(ldsAddr), "v"(gaddr)
               : "memory");
}
__device__ __forceinline__ void waitAsync0() {
  asm volatile("s_wait_asynccnt 0x0" ::: "memory");
}

#define BM   128
#define BN   128
#define ROWB 80                  // 64B of K-data + 16B pad (bank-conflict-free)
#define TILEB (128 * ROWB)       // 10240 B per staged matrix tile
#define BUFB  (2 * TILEB)        // 20480 B per double-buffer stage

// EPI: 0 = compress (tanh, bf16 out, shape (M,N))
//      1 = decompress mid (reshape (M,4096)->(M*8,512), +bf16 resid, tanh, bf16)
//      2 = decompress final (reshape, +f32 resid, tanh, f32 out)
template <int EPI>
__global__ __launch_bounds__(256)
void gemm_tanh_wmma(const __bf16* __restrict__ A,
                    const __bf16* __restrict__ W,
                    const float* __restrict__ bias,
                    const __bf16* __restrict__ residB,
                    const float* __restrict__ residF,
                    __bf16* __restrict__ outB,
                    float* __restrict__ outF,
                    int M, int N, int K) {
  extern __shared__ char smem[];
  const unsigned int ldsBase = (unsigned int)(uintptr_t)smem;

  const int tid  = threadIdx.x;
  const int lane = tid & 31;
  const int wave = tid >> 5;
  const int nBlocks = N >> 7;
  const int bm = blockIdx.x / nBlocks;
  const int bn = blockIdx.x - bm * nBlocks;
  const int mBase = bm << 7;
  const int nBase = bn << 7;

  // ---- loader mapping: thread -> (row, 32B half-row) of the 128x32 slab ----
  const int lr = tid >> 1;                 // 0..127
  const int lcElem = (tid & 1) << 4;       // element offset 0 or 16
  int aRow = mBase + lr; if (aRow > M - 1) aRow = M - 1;   // clamp ragged M
  const __bf16* aSrc = A + (size_t)aRow * K + lcElem;
  const __bf16* bSrc = W + (size_t)(nBase + lr) * K + lcElem;
  const unsigned int ldsRow = (unsigned int)lr * ROWB + ((tid & 1) << 5);

  // ---- compute mapping: 8 waves = 4(M) x 2(N); wave tile 32x64 ----
  const int wm = wave & 3;
  const int wn = wave >> 2;
  const int l16 = lane & 15;
  const int koff2 = (lane >> 4) << 4;      // byte offset of K-half: 0 or 16

  v8f acc[2][4] = {};
  const int ksteps = K >> 5;

  auto issue = [&](int buf, int k0) {
    const unsigned int base = ldsBase + (unsigned int)buf * BUFB + ldsRow;
    const __bf16* ga = aSrc + k0;
    const __bf16* gb = bSrc + k0;
    asyncCopy16(base,              ga);
    asyncCopy16(base + 16,         ga + 8);
    asyncCopy16(base + TILEB,      gb);
    asyncCopy16(base + TILEB + 16, gb + 8);
  };

  issue(0, 0);
  for (int i = 0; i < ksteps; ++i) {
    waitAsync0();
    __syncthreads();                       // all slabs visible; prev compute done
    if (i + 1 < ksteps) issue((i + 1) & 1, (i + 1) << 5);

    const char* base = smem + (i & 1) * BUFB;
    BfPack a[2], b[4];
#pragma unroll
    for (int s = 0; s < 2; ++s) {
      const char* p = base + (wm * 32 + s * 16 + l16) * ROWB + koff2;
      a[s].u[0] = *(const v4u*)p;
      a[s].u[1] = *(const v4u*)(p + 32);
    }
#pragma unroll
    for (int t = 0; t < 4; ++t) {
      const char* p = base + TILEB + (wn * 64 + t * 16 + l16) * ROWB + koff2;
      b[t].u[0] = *(const v4u*)p;
      b[t].u[1] = *(const v4u*)(p + 32);
    }
#pragma unroll
    for (int s = 0; s < 2; ++s)
#pragma unroll
      for (int t = 0; t < 4; ++t)
        acc[s][t] = __builtin_amdgcn_wmma_f32_16x16x32_bf16(
            false, a[s].v, false, b[t].v, (short)0, acc[s][t], false, false);
  }

  // ---- epilogue: C VGPR r -> (row r | r+8, col lane%16) ----
  const int rowAdd = (lane >> 4) << 3;
  float bv[4];
#pragma unroll
  for (int t = 0; t < 4; ++t) bv[t] = bias[nBase + wn * 64 + t * 16 + l16];

#pragma unroll
  for (int s = 0; s < 2; ++s) {
#pragma unroll
    for (int t = 0; t < 4; ++t) {
      const int col = nBase + wn * 64 + t * 16 + l16;
#pragma unroll
      for (int r = 0; r < 8; ++r) {
        const int m = mBase + wm * 32 + s * 16 + r + rowAdd;
        if (m < M) {
          float v = acc[s][t][r] + bv[t];
          if (EPI == 0) {
            outB[(size_t)m * N + col] = f2bf(tanhf(v));
          } else {
            const int orow = (m << 3) + (col >> 9);   // (M,4096)->(M*8,512)
            const int ocol = col & 511;
            const size_t oi = (size_t)orow * 512 + ocol;
            if (EPI == 1) {
              v += bf2f(residB[oi]);
              outB[oi] = f2bf(tanhf(v));
            } else {
              v += residF[oi];
              outF[oi] = tanhf(v);
            }
          }
        }
      }
    }
  }
}

extern "C" void kernel_launch(void* const* d_in, const int* in_sizes, int n_in,
                              void* d_out, int out_size, void* d_ws, size_t ws_size,
                              hipStream_t stream) {
  (void)in_sizes; (void)n_in; (void)out_size; (void)ws_size;
  const float* seq = (const float*)d_in[0];   // (2, 32768, 512)
  const float* Wc  = (const float*)d_in[1];   // (6, 512, 4096)
  const float* bc  = (const float*)d_in[2];   // (6, 512)
  const float* Wd  = (const float*)d_in[3];   // (6, 4096, 512)
  const float* bd  = (const float*)d_in[4];   // (6, 4096)
  float* out = (float*)d_out;                 // (2, 32768, 512)

  const long WELEMS = 6L * 512 * 4096;
  const size_t LDSB = 2 * BUFB;               // 40 KB dynamic LDS

  char* ws = (char*)d_ws;
  size_t off = 0;
  auto carve = [&](size_t bytes) -> char* {
    char* p = ws + off;
    off += (bytes + 255) & ~(size_t)255;
    return p;
  };

  __bf16* WcB = (__bf16*)carve((size_t)WELEMS * 2);
  __bf16* WdB = (__bf16*)carve((size_t)WELEMS * 2);
  const long hierElems[6] = {33554432L, 4194304L, 524288L, 65536L, 8192L, 1024L};
  __bf16* hier[6];
  for (int i = 0; i < 6; ++i) hier[i] = (__bf16*)carve((size_t)hierElems[i] * 2);
  __bf16* ping = (__bf16*)carve(4194304L * 2);
  __bf16* pong = (__bf16*)carve(4194304L * 2);

  // --- f32 -> bf16 staging (weights + input sequence) ---
  f32_to_bf16_kernel<<<4096, 256, 0, stream>>>(Wc, WcB, WELEMS);
  f32_to_bf16_kernel<<<4096, 256, 0, stream>>>(Wd, WdB, WELEMS);
  f32_to_bf16_kernel<<<8192, 256, 0, stream>>>(seq, hier[0], 33554432L);

  // --- compress: 5 levels, A=(M,4096) W=(512,4096) ---
  const int Mc[5] = {8192, 1024, 128, 16, 2};
  for (int l = 0; l < 5; ++l) {
    const int M = Mc[l], N = 512, K = 4096;
    const int blocks = ((M + 127) / 128) * (N / 128);
    gemm_tanh_wmma<0><<<blocks, 256, LDSB, stream>>>(
        hier[l], WcB + (size_t)l * 512 * 4096, bc + (size_t)l * 512,
        nullptr, nullptr, hier[l + 1], nullptr, M, N, K);
  }

  // --- decompress: 5 levels, A=(M,512) W=(4096,512) ---
  const __bf16* cur = hier[5];
  __bf16* bufs[2] = {ping, pong};
  for (int i = 0; i < 5; ++i) {
    const int l = 4 - i;
    int M = 2;
    for (int j = 0; j < 4 - l; ++j) M *= 8;   // 2 * 8^(4-l)
    const int N = 4096, K = 512;
    const int blocks = ((M + 127) / 128) * (N / 128);
    if (l > 0) {
      gemm_tanh_wmma<1><<<blocks, 256, LDSB, stream>>>(
          cur, WdB + (size_t)l * 4096 * 512, bd + (size_t)l * 4096,
          hier[l], nullptr, bufs[i & 1], nullptr, M, N, K);
      cur = bufs[i & 1];
    } else {
      gemm_tanh_wmma<2><<<blocks, 256, LDSB, stream>>>(
          cur, WdB, bd, nullptr, seq, nullptr, out, M, N, K);
    }
  }
}
